// GeBertGATv2Encoder_52252572123581
// MI455X (gfx1250) — compile-verified, matrix operand order checked by reading, not run on hardware
//
#include <hip/hip_runtime.h>
#include <hip/hip_bf16.h>

#define N_NODES   100000
#define N_EDGES   800000
#define DIM       128
#define HEADS     4
#define DH        32
#define N_TRG     50000
#define NEG_SLOPE 0.2f

typedef __attribute__((ext_vector_type(16))) __bf16        v16bf;
typedef __attribute__((ext_vector_type(8)))  float         v8f;
typedef __attribute__((ext_vector_type(4)))  unsigned int  u32x4;
typedef __attribute__((ext_vector_type(8)))  unsigned int  u32x8;

// ---------- helpers ----------
static __device__ __forceinline__ __bf16 f2bf(float f) {
    unsigned u = __float_as_uint(f);
    unsigned r = u + 0x7FFFu + ((u >> 16) & 1u);        // round-to-nearest-even
    unsigned short hs = (unsigned short)(r >> 16);
    return __builtin_bit_cast(__bf16, hs);
}
// order-preserving float -> uint encoding (for atomicMax-based segment max)
static __device__ __forceinline__ unsigned encf(float f) {
    unsigned u = __float_as_uint(f);
    return (u & 0x80000000u) ? ~u : (u | 0x80000000u);
}
static __device__ __forceinline__ float decf(unsigned u) {
    unsigned v = (u & 0x80000000u) ? (u & 0x7FFFFFFFu) : ~u;
    return __uint_as_float(v);
}

// ---------- weight repack into WMMA B-fragment layout (bf16) ----------
// packed index = ((mat*8 + nt)*4 + kt)*32*16 + lane*16 + i
// lane holds column n = nt*16 + (lane&15); K base kb = kt*32 + (lane<16 ? 0 : 8)
// halves 0..7 -> K = kb..kb+7 ; halves 8..15 -> K = kb+16..kb+23
__global__ __launch_bounds__(256)
void pack_weights(const float* __restrict__ Ws, const float* __restrict__ loW,
                  __bf16* __restrict__ pW) {
    int t = blockIdx.x * 256 + threadIdx.x;
    if (t >= 5 * 8 * 4 * 32) return;
    int lane = t & 31;
    int kt   = (t >> 5) & 3;
    int nt   = (t >> 7) & 7;
    int mat  = t >> 10;
    const float* W = (mat < 4) ? (Ws + (size_t)mat * DIM * DIM) : loW;
    int n  = nt * 16 + (lane & 15);
    int kb = kt * 32 + ((lane < 16) ? 0 : 8);
    __bf16* d = pW + (size_t)t * 16;
#pragma unroll
    for (int i = 0; i < 8; ++i)  d[i]     = f2bf(W[(kb + i) * DIM + n]);
#pragma unroll
    for (int i = 0; i < 8; ++i)  d[8 + i] = f2bf(W[(kb + 16 + i) * DIM + n]);
}

// ---------- node GEMM:  Y = X @ W + bias ----------
// TDM stages the 32 KB packed weight matrix into LDS (one DMA per workgroup,
// overlapped with per-lane A-fragment builds); WMMA inner loop reads B from LDS.
__global__ __launch_bounds__(256)
void gemm_node(const float* __restrict__ X, const __bf16* __restrict__ pW,
               const float* __restrict__ bias, float* __restrict__ Y) {
    __shared__ __bf16 smW[8 * 4 * 32 * 16];   // 16384 bf16 = 32 KB

    int wave = (blockIdx.x * 256 + threadIdx.x) >> 5;
    int lane = threadIdx.x & 31;
    int row0 = wave * 16;

    // ---- wave 0: kick off Tensor-DMA of packed W -> LDS (8192 dwords, 1-D)
    if (threadIdx.x < 32) {
        unsigned           lds_off = (unsigned)(size_t)(void*)&smW[0];
        unsigned long long ga      = (unsigned long long)(const void*)pW;
        u32x4 g0;
        g0[0] = 1u;                                          // count=1 (valid D#)
        g0[1] = lds_off;                                     // lds_addr
        g0[2] = (unsigned)ga;                                // global_addr[31:0]
        g0[3] = (unsigned)((ga >> 32) & 0x1FFFFFFu) | (2u << 30); // addr[56:32] | type=2
        u32x8 g1;
        g1[0] = 0x00020000u;   // workgroup_mask=0, data_size=2 (4 B)
        g1[1] = 0x20000000u;   // tensor_dim0 = 8192 (bits[79:48] -> w1[31:16])
        g1[2] = 0u;            // tensor_dim0 hi / tensor_dim1 lo
        g1[3] = 0x20000000u;   // tile_dim0 = 8192 (bits[127:112])
        g1[4] = 0u;            // tile_dim1 / tile_dim2 unused
        g1[5] = 0x2000u;       // tensor_dim0_stride = 8192
        g1[6] = 0u;
        g1[7] = 0u;
        asm volatile("tensor_load_to_lds %0, %1" :: "s"(g0), "s"(g1) : "memory");
    }

    if (row0 < N_NODES) {
        int m  = row0 + (lane & 15);
        int kb = (lane < 16) ? 0 : 8;
        const float* xr = X + (size_t)m * DIM;

        // ---- build A fragments (overlaps with the TDM transfer)
        v16bf a[4];
#pragma unroll
        for (int kt = 0; kt < 4; ++kt) {
            int k = kt * 32 + kb;
            float4 f0 = *(const float4*)(xr + k);
            float4 f1 = *(const float4*)(xr + k + 4);
            float4 f2 = *(const float4*)(xr + k + 16);
            float4 f3 = *(const float4*)(xr + k + 20);
            a[kt][0]  = f2bf(f0.x); a[kt][1]  = f2bf(f0.y); a[kt][2]  = f2bf(f0.z); a[kt][3]  = f2bf(f0.w);
            a[kt][4]  = f2bf(f1.x); a[kt][5]  = f2bf(f1.y); a[kt][6]  = f2bf(f1.z); a[kt][7]  = f2bf(f1.w);
            a[kt][8]  = f2bf(f2.x); a[kt][9]  = f2bf(f2.y); a[kt][10] = f2bf(f2.z); a[kt][11] = f2bf(f2.w);
            a[kt][12] = f2bf(f3.x); a[kt][13] = f2bf(f3.y); a[kt][14] = f2bf(f3.z); a[kt][15] = f2bf(f3.w);
        }

        if (threadIdx.x < 32) __builtin_amdgcn_s_wait_tensorcnt(0);
        __syncthreads();

#pragma unroll
        for (int nt = 0; nt < 8; ++nt) {
            float b0 = bias[nt * 16 + (lane & 15)];
            v8f acc;
#pragma unroll
            for (int i = 0; i < 8; ++i) acc[i] = b0;
#pragma unroll
            for (int kt = 0; kt < 4; ++kt) {
                v16bf bf = *(const v16bf*)(smW + (((nt * 4 + kt) * 32 + lane) << 4));
                acc = __builtin_amdgcn_wmma_f32_16x16x32_bf16(
                    false, a[kt], false, bf, (short)0, acc, false, false);
            }
            int col = nt * 16 + (lane & 15);
            int rb  = row0 + ((lane >> 4) << 3);
#pragma unroll
            for (int r = 0; r < 8; ++r)
                Y[(size_t)(rb + r) * DIM + col] = acc[r];
        }
    } else {
        if (threadIdx.x < 32) __builtin_amdgcn_s_wait_tensorcnt(0);
        __syncthreads();
    }
}

// ---------- per-layer init: zero aggregation buffer, max/sum accumulators ----------
__global__ __launch_bounds__(256)
void init_layer(float* __restrict__ out, unsigned* __restrict__ menc,
                float* __restrict__ ssum) {
    int i = blockIdx.x * 256 + threadIdx.x;
    if (i < N_NODES * DIM) out[i] = 0.0f;
    if (i < N_NODES * HEADS) { menc[i] = 0u; ssum[i] = 0.0f; }
}

// ---------- edge pass A: scores + segment max (wave per edge) ----------
__global__ __launch_bounds__(256)
void edge_score_max(const float* __restrict__ h, const int* __restrict__ src,
                    const int* __restrict__ dst, const float* __restrict__ att,
                    float* __restrict__ scores, unsigned* __restrict__ menc) {
    int e    = (blockIdx.x * 256 + threadIdx.x) >> 5;
    int lane = threadIdx.x & 31;
    if (e >= N_EDGES) return;
    int s = src[e], d = dst[e];
    float4 a = *(const float4*)(h + (size_t)s * DIM + lane * 4);
    float4 b = *(const float4*)(h + (size_t)d * DIM + lane * 4);
    float4 w = *(const float4*)(att + lane * 4);
    auto lr = [](float v) { return v > 0.0f ? v : NEG_SLOPE * v; };
    float p = lr(a.x + b.x) * w.x + lr(a.y + b.y) * w.y +
              lr(a.z + b.z) * w.z + lr(a.w + b.w) * w.w;
    // reduce the 8 lanes of each head (lanes h*8 .. h*8+7)
    p += __shfl_xor(p, 1, 32);
    p += __shfl_xor(p, 2, 32);
    p += __shfl_xor(p, 4, 32);
    int hh = lane >> 3;
    if ((lane & 7) == 0) {
        scores[(size_t)e * HEADS + hh] = p;
        atomicMax(menc + (size_t)d * HEADS + hh, encf(p));
    }
}

// ---------- edge pass B: a = exp(score - max); segment sum ----------
__global__ __launch_bounds__(256)
void edge_exp_sum(float* __restrict__ scores, const int* __restrict__ dst,
                  const unsigned* __restrict__ menc, float* __restrict__ ssum) {
    int t = blockIdx.x * 256 + threadIdx.x;
    if (t >= N_EDGES * HEADS) return;
    int e = t >> 2, hh = t & 3;
    int d = dst[e];
    float m = decf(menc[(size_t)d * HEADS + hh]);
    float a = __expf(scores[t] - m);
    scores[t] = a;
    atomicAdd(ssum + (size_t)d * HEADS + hh, a);
}

// ---------- edge pass C: out[dst] += alpha * h[src]  (wave per edge) ----------
__global__ __launch_bounds__(256)
void edge_aggregate(const float* __restrict__ h, const int* __restrict__ src,
                    const int* __restrict__ dst, const float* __restrict__ scores,
                    const float* __restrict__ ssum, float* __restrict__ out) {
    int e    = (blockIdx.x * 256 + threadIdx.x) >> 5;
    int lane = threadIdx.x & 31;
    if (e >= N_EDGES) return;
    int s = src[e], d = dst[e];
    int hh = lane >> 3;
    float alpha = scores[(size_t)e * HEADS + hh] / ssum[(size_t)d * HEADS + hh];
    float4 v = *(const float4*)(h + (size_t)s * DIM + lane * 4);
    float* o = out + (size_t)d * DIM + lane * 4;
    atomicAdd(o + 0, v.x * alpha);
    atomicAdd(o + 1, v.y * alpha);
    atomicAdd(o + 2, v.z * alpha);
    atomicAdd(o + 3, v.w * alpha);
}

// ---------- bias + exact GELU ----------
__global__ __launch_bounds__(256)
void bias_act(float* __restrict__ Y, const float* __restrict__ bout, int do_gelu) {
    int i = blockIdx.x * 256 + threadIdx.x;
    if (i >= N_NODES * DIM) return;
    float x = Y[i] + bout[i & (DIM - 1)];
    if (do_gelu) x = 0.5f * x * (1.0f + erff(x * 0.70710678118654752f));
    Y[i] = x;
}

// ---------- fused head: WMMA GEMM + bias + LayerNorm ----------
__global__ __launch_bounds__(256)
void head_gemm_ln(const float* __restrict__ X, const __bf16* __restrict__ pW,
                  const float* __restrict__ bias, const float* __restrict__ gamma,
                  const float* __restrict__ beta, float* __restrict__ out) {
    __shared__ float tile[16][132];
    __shared__ float psum[16][16];
    __shared__ float psq[16][16];
    int nt   = threadIdx.x >> 5;          // wave id = n-tile
    int lane = threadIdx.x & 31;
    int row0 = blockIdx.x * 16;
    int m  = row0 + (lane & 15);
    int kb = (lane < 16) ? 0 : 8;
    const float* xr = X + (size_t)m * DIM;

    v16bf a[4];
#pragma unroll
    for (int kt = 0; kt < 4; ++kt) {
        int k = kt * 32 + kb;
        float4 f0 = *(const float4*)(xr + k);
        float4 f1 = *(const float4*)(xr + k + 4);
        float4 f2 = *(const float4*)(xr + k + 16);
        float4 f3 = *(const float4*)(xr + k + 20);
        a[kt][0]  = f2bf(f0.x); a[kt][1]  = f2bf(f0.y); a[kt][2]  = f2bf(f0.z); a[kt][3]  = f2bf(f0.w);
        a[kt][4]  = f2bf(f1.x); a[kt][5]  = f2bf(f1.y); a[kt][6]  = f2bf(f1.z); a[kt][7]  = f2bf(f1.w);
        a[kt][8]  = f2bf(f2.x); a[kt][9]  = f2bf(f2.y); a[kt][10] = f2bf(f2.z); a[kt][11] = f2bf(f2.w);
        a[kt][12] = f2bf(f3.x); a[kt][13] = f2bf(f3.y); a[kt][14] = f2bf(f3.z); a[kt][15] = f2bf(f3.w);
    }

    float b0 = bias[nt * 16 + (lane & 15)];
    v8f acc;
#pragma unroll
    for (int i = 0; i < 8; ++i) acc[i] = b0;
#pragma unroll
    for (int kt = 0; kt < 4; ++kt) {
        v16bf bf = *(const v16bf*)(pW + (((size_t)nt * 4 + kt) * 32 + lane) * 16);
        acc = __builtin_amdgcn_wmma_f32_16x16x32_bf16(
            false, a[kt], false, bf, (short)0, acc, false, false);
    }
    int col = nt * 16 + (lane & 15);
    int rb  = (lane >> 4) << 3;
#pragma unroll
    for (int r = 0; r < 8; ++r) tile[rb + r][col] = acc[r];
    __syncthreads();

    // LayerNorm over the 128 columns of each of the 16 rows
    int r = threadIdx.x >> 4;
    int g = threadIdx.x & 15;
    float su = 0.0f, sq = 0.0f;
#pragma unroll
    for (int j = 0; j < 8; ++j) {
        float v = tile[r][g * 8 + j];
        su += v; sq += v * v;
    }
    psum[r][g] = su; psq[r][g] = sq;
    __syncthreads();
    float tot = 0.0f, tq = 0.0f;
#pragma unroll
    for (int j = 0; j < 16; ++j) { tot += psum[r][j]; tq += psq[r][j]; }
    float mu  = tot * (1.0f / DIM);
    float var = tq * (1.0f / DIM) - mu * mu;
    float rs  = rsqrtf(var + 1e-12f);
#pragma unroll
    for (int j = 0; j < 8; ++j) {
        int c = g * 8 + j;
        out[(size_t)(row0 + r) * DIM + c] =
            (tile[r][c] - mu) * rs * gamma[c] + beta[c];
    }
}

// ---------- launcher ----------
extern "C" void kernel_launch(void* const* d_in, const int* in_sizes, int n_in,
                              void* d_out, int out_size, void* d_ws, size_t ws_size,
                              hipStream_t stream) {
    const float* x     = (const float*)d_in[0];
    const int*   ei    = (const int*)  d_in[1];
    const float* Ws    = (const float*)d_in[3];
    const float* bWs   = (const float*)d_in[4];
    const float* atts  = (const float*)d_in[5];
    const float* bouts = (const float*)d_in[6];
    const float* loW   = (const float*)d_in[7];
    const float* lob   = (const float*)d_in[8];
    const float* lng   = (const float*)d_in[9];
    const float* lnb   = (const float*)d_in[10];
    const int* src = ei;
    const int* dst = ei + N_EDGES;

    char* ws = (char*)d_ws;
    float*    bufA   = (float*)   (ws);                       // 51.2 MB: h (post-GEMM)
    float*    bufB   = (float*)   (ws + 51200000);            // 51.2 MB: aggregation / x_next
    float*    scores = (float*)   (ws + 102400000);           // 12.8 MB: per-edge per-head
    unsigned* menc   = (unsigned*)(ws + 115200000);           // 1.6 MB: encoded segment max
    float*    ssum   = (float*)   (ws + 116800000);           // 1.6 MB: segment sum
    __bf16*   pW     = (__bf16*)  (ws + 118400000);           // 160 KB: packed weights

    pack_weights<<<20, 256, 0, stream>>>(Ws, loW, pW);

    const int gemm_blocks  = (N_NODES / 16 + 7) / 8;          // 782
    const int node_blocks  = (N_NODES * DIM) / 256;           // 50000
    const int edgeW_blocks = N_EDGES / 8;                     // 100000 (wave per edge)
    const int edgeT_blocks = (N_EDGES * HEADS) / 256;         // 12500

    const float* xcur = x;
    for (int l = 0; l < 4; ++l) {
        gemm_node<<<gemm_blocks, 256, 0, stream>>>(
            xcur, pW + (size_t)l * 16384, bWs + l * DIM, bufA);
        init_layer<<<node_blocks, 256, 0, stream>>>(bufB, menc, ssum);
        edge_score_max<<<edgeW_blocks, 256, 0, stream>>>(
            bufA, src, dst, atts + l * DIM, scores, menc);
        edge_exp_sum<<<edgeT_blocks, 256, 0, stream>>>(scores, dst, menc, ssum);
        edge_aggregate<<<edgeW_blocks, 256, 0, stream>>>(
            bufA, src, dst, scores, ssum, bufB);
        bias_act<<<node_blocks, 256, 0, stream>>>(bufB, bouts + l * DIM, l < 3 ? 1 : 0);
        xcur = bufB;
    }

    head_gemm_ln<<<N_TRG / 16, 256, 0, stream>>>(
        bufB, pW + (size_t)4 * 16384, lob, lng, lnb, (float*)d_out);
}
